// model_82695300317475
// MI455X (gfx1250) — matrix operand order repaired and checked by reading, not compiled, not run on hardware
//
#include <hip/hip_runtime.h>
#include <hip/hip_bf16.h>

typedef __attribute__((ext_vector_type(16))) _Float16 v16h;
typedef __attribute__((ext_vector_type(8)))  _Float16 v8h;
typedef __attribute__((ext_vector_type(8)))  float    v8f;

#define FLAG_ACC  1
#define FLAG_TANH 2

static constexpr int B_  = 16;
static constexpr int S_  = 128;
static constexpr int TV_ = 20;
static constexpr int TW_ = 100;
static constexpr int H_  = 1024;
static constexpr int H2_ = 512;
static constexpr int L_  = 2;
static constexpr int V_  = 50000;
static constexpr int VV_ = 5000;

// ---------------------------------------------------------------------------
// C[M,N] = A[M,K](f16 compact, lda=K) * B[N,K]^T (f16 row-major) [+C][+bias][tanh]
// Each wave computes TWO adjacent 16x16 tiles (shared A fragment, 2 accums).
// K-loop steps 32 via v_wmma_f32_16x16x32_f16.
// A frag (16-bit A 16x32 layout): lanes 0-15 K runs {kk..kk+7, kk+16..kk+23},
// lanes 16-31 runs {kk+8..kk+15, kk+24..kk+31}. B frag: same K runs as A;
// lane&15 selects N column == W row.
// ---------------------------------------------------------------------------
__global__ __launch_bounds__(256) void k_gemm(
    const _Float16* __restrict__ Ah,
    const _Float16* __restrict__ Bw,
    const float* __restrict__ bias,
    float* __restrict__ C, size_t ldc,
    int M, int N, int K, int flags)
{
  int lane = threadIdx.x & 31;
  int wave = threadIdx.x >> 5;
  int n0 = (blockIdx.x * 8 + wave) * 32;   // two 16-wide tiles per wave
  int m0 = blockIdx.y * 16;
  if (n0 >= N || m0 >= M) return;
  int l  = lane & 15;
  int hf = lane >> 4;
  int ar = m0 + l; if (ar >= M) ar = M - 1;
  const _Float16* Arow = Ah + (size_t)ar * K;
  int br0 = n0 + l;
  int br1 = n0 + 16 + l;
  const _Float16* Brow0 = Bw + (size_t)(br0 < N ? br0 : 0) * K;
  const _Float16* Brow1 = Bw + (size_t)(br1 < N ? br1 : 0) * K;

  v8f acc0 = {};
  v8f acc1 = {};
  for (int kk = 0; kk < K; kk += 32) {
    int ka = kk + hf * 8;
    const v8h a0 = *(const v8h*)(Arow + ka);
    const v8h a1 = *(const v8h*)(Arow + ka + 16);
    v16h af = __builtin_shufflevector(a0, a1, 0,1,2,3,4,5,6,7,8,9,10,11,12,13,14,15);
    int kb = kk + hf * 8;
    const v8h b00 = *(const v8h*)(Brow0 + kb);
    const v8h b01 = *(const v8h*)(Brow0 + kb + 16);
    const v8h b10 = *(const v8h*)(Brow1 + kb);
    const v8h b11 = *(const v8h*)(Brow1 + kb + 16);
    v16h bf0 = __builtin_shufflevector(b00, b01, 0,1,2,3,4,5,6,7,8,9,10,11,12,13,14,15);
    v16h bf1 = __builtin_shufflevector(b10, b11, 0,1,2,3,4,5,6,7,8,9,10,11,12,13,14,15);
    acc0 = __builtin_amdgcn_wmma_f32_16x16x32_f16(false, af, false, bf0,
                                                  (short)0, acc0, false, false);
    acc1 = __builtin_amdgcn_wmma_f32_16x16x32_f16(false, af, false, bf1,
                                                  (short)0, acc1, false, false);
  }
  for (int tile = 0; tile < 2; ++tile) {
    int col = n0 + tile * 16 + l;
    if (col >= N) continue;
    const v8f& acc = tile ? acc1 : acc0;
    float bb = bias ? bias[col] : 0.0f;
    for (int v = 0; v < 8; ++v) {
      int row = m0 + hf * 8 + v;
      if (row < M) {
        size_t off = (size_t)row * ldc + col;
        float r = acc[v] + bb;
        if (flags & FLAG_ACC)  r += C[off];
        if (flags & FLAG_TANH) r = tanhf(r);
        C[off] = r;
      }
    }
  }
}

// f32 -> f16, contiguous, 8 elems/thread (n must be multiple of 8; all weights are)
__global__ void k_cvt_f16(const float* __restrict__ s, _Float16* __restrict__ d, size_t n) {
  size_t i  = (size_t)blockIdx.x * blockDim.x + threadIdx.x;
  size_t st = (size_t)gridDim.x * blockDim.x;
  size_t n8 = n >> 3;
  for (; i < n8; i += st) {
    const float4 a0 = *(const float4*)(s + i * 8);
    const float4 a1 = *(const float4*)(s + i * 8 + 4);
    v8h o;
    o[0] = (_Float16)a0.x; o[1] = (_Float16)a0.y; o[2] = (_Float16)a0.z; o[3] = (_Float16)a0.w;
    o[4] = (_Float16)a1.x; o[5] = (_Float16)a1.y; o[6] = (_Float16)a1.z; o[7] = (_Float16)a1.w;
    *(v8h*)(d + i * 8) = o;
  }
}

// strided f32 (rows x cols, lda) -> compact f16 (lda=cols); cols multiple of 8
__global__ void k_cvt2d_f16(const float* __restrict__ src, int lda,
                            _Float16* __restrict__ dst, int rows, int cols) {
  int i = blockIdx.x * blockDim.x + threadIdx.x;
  int cpr = cols >> 3;
  if (i >= rows * cpr) return;
  int r = i / cpr, c = (i % cpr) << 3;
  const float4 a0 = *(const float4*)(src + (size_t)r * lda + c);
  const float4 a1 = *(const float4*)(src + (size_t)r * lda + c + 4);
  v8h o;
  o[0] = (_Float16)a0.x; o[1] = (_Float16)a0.y; o[2] = (_Float16)a0.z; o[3] = (_Float16)a0.w;
  o[4] = (_Float16)a1.x; o[5] = (_Float16)a1.y; o[6] = (_Float16)a1.z; o[7] = (_Float16)a1.w;
  *(v8h*)(dst + (size_t)r * cols + c) = o;
}

__global__ void k_fill(float* __restrict__ p, float v, int n) {
  int i = blockIdx.x * blockDim.x + threadIdx.x;
  if (i < n) p[i] = v;
}

// embedding gather, float4 per thread (H multiple of 4)
__global__ void k_gather(const float* __restrict__ emb, const int* __restrict__ idx,
                         float* __restrict__ out, int rows, int H) {
  int i = blockIdx.x * blockDim.x + threadIdx.x;
  int cpr = H >> 2;
  if (i >= rows * cpr) return;
  int r = i / cpr, c = (i % cpr) << 2;
  *(float4*)(out + (size_t)r * H + c) =
      *(const float4*)(emb + (size_t)idx[r] * H + c);
}

__global__ void k_shift(const int* __restrict__ tok, int* __restrict__ prev, int Bt, int T) {
  int i = blockIdx.x * blockDim.x + threadIdx.x;
  if (i >= Bt * T) return;
  prev[i] = (i % T == 0) ? 3 : tok[i - 1];   // START token = 3
}

__global__ void k_copy2d(const float* __restrict__ src, size_t sld,
                         float* __restrict__ dst, size_t dld, int rows, int cols) {
  int i = blockIdx.x * blockDim.x + threadIdx.x;
  if (i >= rows * cols) return;
  int r = i / cols, c = i % cols;
  dst[(size_t)r * dld + c] = src[(size_t)r * sld + c];
}

// gates (Bn, 4*Hd) with row stride gld; order i,f,g,o. c updated in place.
__global__ void k_lstm_cell(const float* __restrict__ g, size_t gld,
                            const float* __restrict__ cin, float* __restrict__ cout,
                            float* __restrict__ hout, size_t hld, int Bn, int Hd) {
  int i = blockIdx.x * blockDim.x + threadIdx.x;
  if (i >= Bn * Hd) return;
  int b = i / Hd, j = i % Hd;
  const float* gr = g + (size_t)b * gld;
  float ig = 1.0f / (1.0f + __expf(-gr[j]));
  float fg = 1.0f / (1.0f + __expf(-gr[Hd + j]));
  float gg = tanhf(gr[2 * Hd + j]);
  float og = 1.0f / (1.0f + __expf(-gr[3 * Hd + j]));
  float c2 = fg * cin[i] + ig * gg;
  cout[i] = c2;
  hout[(size_t)b * hld + j] = og * tanhf(c2);
}

// out[b,t,s] = dot(keys[b,s,:], q[b,t,:])  — one wave per output element
__global__ void k_scores(const float* __restrict__ keys, const float* __restrict__ q,
                         float* __restrict__ out, int Bq, int Tq, int Sk, int H) {
  int w = (int)((blockIdx.x * blockDim.x + threadIdx.x) >> 5);
  int lane = threadIdx.x & 31;
  if (w >= Bq * Tq * Sk) return;
  int s = w % Sk, t = (w / Sk) % Tq, b = w / (Sk * Tq);
  const float* kr = keys + ((size_t)b * Sk + s) * H;
  const float* qr = q + ((size_t)b * Tq + t) * H;
  float acc = 0.0f;
  for (int j = lane; j < H; j += 32) acc += kr[j] * qr[j];
  for (int o = 16; o; o >>= 1) acc += __shfl_xor(acc, o, 32);
  if (lane == 0) out[w] = acc;
}

__global__ void k_softmax(float* __restrict__ x, int rows, int n) {
  int w = (int)((blockIdx.x * blockDim.x + threadIdx.x) >> 5);
  int lane = threadIdx.x & 31;
  if (w >= rows) return;
  float* r = x + (size_t)w * n;
  float m = -3.4e38f;
  for (int j = lane; j < n; j += 32) m = fmaxf(m, r[j]);
  for (int o = 16; o; o >>= 1) m = fmaxf(m, __shfl_xor(m, o, 32));
  float sum = 0.0f;
  for (int j = lane; j < n; j += 32) sum += __expf(r[j] - m);
  for (int o = 16; o; o >>= 1) sum += __shfl_xor(sum, o, 32);
  float inv = 1.0f / sum;
  for (int j = lane; j < n; j += 32) r[j] = __expf(r[j] - m) * inv;
}

// out[b,t,h] = sum_s w[b,t,s] * keys[b,s,h]
__global__ void k_ctx(const float* __restrict__ wgt, const float* __restrict__ keys,
                      float* __restrict__ out, int Bq, int Tq, int Sk, int H) {
  int i = blockIdx.x * blockDim.x + threadIdx.x;
  if (i >= Bq * Tq * H) return;
  int hc = i % H, t = (i / H) % Tq, b = i / (H * Tq);
  const float* wr = wgt + ((size_t)b * Tq + t) * Sk;
  const float* kb = keys + (size_t)b * Sk * H + hc;
  float acc = 0.0f;
  for (int s = 0; s < Sk; ++s) acc += wr[s] * kb[(size_t)s * H];
  out[i] = acc;
}

__global__ void k_argmax(const float* __restrict__ x, int rows, int n, int* __restrict__ idx) {
  int w = (int)((blockIdx.x * blockDim.x + threadIdx.x) >> 5);
  int lane = threadIdx.x & 31;
  if (w >= rows) return;
  const float* r = x + (size_t)w * n;
  float bm = -3.4e38f; int bi = 0;
  for (int j = lane; j < n; j += 32) {
    float v = r[j];
    if (v > bm) { bm = v; bi = j; }
  }
  for (int o = 16; o; o >>= 1) {
    float om = __shfl_xor(bm, o, 32);
    int   oi = __shfl_xor(bi, o, 32);
    if (om > bm || (om == bm && oi < bi)) { bm = om; bi = oi; }
  }
  if (lane == 0) idx[w] = bi;
}

// ---------------------------------------------------------------------------
extern "C" void kernel_launch(void* const* d_in, const int* in_sizes, int n_in,
                              void* d_out, int out_size, void* d_ws, size_t ws_size,
                              hipStream_t stream) {
  (void)in_sizes; (void)n_in; (void)out_size;
  const int*   t_inp    = (const int*)d_in[0];
  const int*   t_out    = (const int*)d_in[1];
  const int*   t_verbs  = (const int*)d_in[2];
  const float* encemb   = (const float*)d_in[3];
  const float* decemb   = (const float*)d_in[4];
  const float* vemb     = (const float*)d_in[5];
  const float* enc_Wih  = (const float*)d_in[6];
  const float* enc_Whh  = (const float*)d_in[7];
  const float* enc_b    = (const float*)d_in[8];
  const float* vdec_Wih = (const float*)d_in[9];
  const float* vdec_Whh = (const float*)d_in[10];
  const float* vdec_b   = (const float*)d_in[11];
  const float* dec_Wih0 = (const float*)d_in[12];
  const float* dec_Wih1 = (const float*)d_in[13];
  const float* dec_Whh  = (const float*)d_in[14];
  const float* dec_b    = (const float*)d_in[15];
  const float* vlinin   = (const float*)d_in[16];
  const float* vlinout  = (const float*)d_in[17];
  const float* dvlinin  = (const float*)d_in[18];
  const float* linout   = (const float*)d_in[19];
  const float* gen_w    = (const float*)d_in[20];
  const float* gen_b    = (const float*)d_in[21];
  const float* vgen_w   = (const float*)d_in[22];
  const float* vgen_b   = (const float*)d_in[23];
  float* outp = (float*)d_out;

  // ---- bump allocator over d_ws ----
  char* wsBase = (char*)d_ws;
  size_t cur = 0;
  auto allocB = [&](size_t bytes) -> void* {
    size_t p = (cur + 255) & ~(size_t)255;
    cur = p + bytes;
    return (void*)(wsBase + p);
  };
  auto allocF = [&](size_t n) { return (float*)allocB(n * 4); };
  auto allocH = [&](size_t n) { return (_Float16*)allocB(n * 2); };
  auto allocI = [&](size_t n) { return (int*)allocB(n * 4); };

  // f16 weight copies (converted once per launch; deterministic)
  _Float16* h_encWih  = allocH((size_t)L_ * 2 * (4 * H2_) * H_);
  _Float16* h_encWhh  = allocH((size_t)L_ * 2 * (4 * H2_) * H2_);
  _Float16* h_vdecWih = allocH((size_t)L_ * (4 * H_) * H_);
  _Float16* h_vdecWhh = allocH((size_t)L_ * (4 * H_) * H_);
  _Float16* h_decWih0 = allocH((size_t)(4 * H_) * (2 * H_));
  _Float16* h_decWih1 = allocH((size_t)(4 * H_) * H_);
  _Float16* h_decWhh  = allocH((size_t)L_ * (4 * H_) * H_);
  _Float16* h_vlinin  = allocH((size_t)H_ * H_);
  _Float16* h_vlinout = allocH((size_t)H_ * 2 * H_);
  _Float16* h_dvlinin = allocH((size_t)H_ * H_);
  _Float16* h_linout  = allocH((size_t)H_ * 2 * H_);
  _Float16* h_gen     = allocH((size_t)V_ * H_);
  _Float16* h_vgen    = allocH((size_t)VV_ * H_);
  // f16 staging for GEMM A operands (max M*K = 2048*1024)
  _Float16* aStage    = allocH((size_t)B_ * S_ * H_);

  // f32 activation scratch
  float* xA      = allocF((size_t)B_ * S_ * H_);
  float* xB      = allocF((size_t)B_ * S_ * H_);
  float* pre     = allocF((size_t)B_ * S_ * (4 * H2_)); // also fits (B*Tv,4H)
  float* cbuf    = allocF((size_t)B_ * H_);
  float* hzero   = allocF((size_t)B_ * H_);
  float* h0s     = allocF((size_t)L_ * B_ * H_);
  float* c0s     = allocF((size_t)L_ * B_ * H_);
  float* vembBuf = allocF((size_t)B_ * TV_ * H_);
  float* vyA     = allocF((size_t)B_ * TV_ * H_);
  float* vyB     = allocF((size_t)B_ * TV_ * H_);
  float* vyC     = allocF((size_t)B_ * TV_ * H_);
  float* qbuf    = allocF((size_t)B_ * TV_ * H_);
  float* watt    = allocF((size_t)B_ * TV_ * S_);
  float* ccb     = allocF((size_t)B_ * TV_ * H_);
  float* catb    = allocF((size_t)B_ * TV_ * 2 * H_);
  float* opv     = allocF((size_t)B_ * TV_ * H_);
  float* vout    = allocF((size_t)B_ * TV_ * VV_);
  int*   vidxs   = allocI((size_t)B_ * TV_);
  int*   tokbuf  = allocI((size_t)B_ * TW_);
  float* demb    = allocF((size_t)B_ * TW_ * H_);
  float* xcat    = allocF((size_t)B_ * 2 * H_);
  float* gbuf    = allocF((size_t)B_ * 4 * H_);
  float* hw0     = allocF((size_t)B_ * H_);
  float* hw1     = allocF((size_t)B_ * H_);
  float* cw0     = allocF((size_t)B_ * H_);
  float* cw1     = allocF((size_t)B_ * H_);
  float* opb     = allocF((size_t)B_ * H_);
  float* vq      = allocF((size_t)B_ * H_);
  float* vw      = allocF((size_t)B_ * TV_);
  float* vcc     = allocF((size_t)B_ * H_);
  float* cat2    = allocF((size_t)B_ * 2 * H_);
  if (cur > ws_size) return;  // workspace too small: launch nothing

  // ---- launch helpers ----
  auto cvt = [&](const float* s, _Float16* d, size_t n) {
    unsigned blocks = (unsigned)((n / 8 + 255) / 256);
    if (blocks > 4096) blocks = 4096;
    k_cvt_f16<<<blocks, 256, 0, stream>>>(s, d, n);
  };
  // stage A (f32, strided) -> f16 compact, then WMMA GEMM
  auto gemm = [&](const float* A, int lda, const _Float16* Bw, const float* bias,
                  float* C, size_t ldc, int M, int N, int K, int flags) {
    int n8 = M * (K >> 3);
    k_cvt2d_f16<<<(n8 + 255) / 256, 256, 0, stream>>>(A, lda, aStage, M, K);
    dim3 grid((unsigned)(((N + 31) / 32 + 7) / 8), (unsigned)((M + 15) / 16));
    k_gemm<<<grid, 256, 0, stream>>>(aStage, Bw, bias, C, ldc, M, N, K, flags);
  };
  auto copy2d = [&](const float* src, size_t sld, float* dst, size_t dld, int rows, int cols) {
    int n = rows * cols;
    k_copy2d<<<(n + 255) / 256, 256, 0, stream>>>(src, sld, dst, dld, rows, cols);
  };
  auto fillz = [&](float* p, int n) {
    k_fill<<<(n + 255) / 256, 256, 0, stream>>>(p, 0.0f, n);
  };
  auto gather = [&](const float* emb, const int* idx, float* dst, int rows) {
    int n = rows * (H_ >> 2);
    k_gather<<<(n + 255) / 256, 256, 0, stream>>>(emb, idx, dst, rows, H_);
  };
  auto cell = [&](const float* g, size_t gld, float* c, float* hout, size_t hld, int Hd) {
    int n = B_ * Hd;
    k_lstm_cell<<<(n + 255) / 256, 256, 0, stream>>>(g, gld, c, c, hout, hld, B_, Hd);
  };

  // ---- weight precision conversion ----
  cvt(enc_Wih,  h_encWih,  (size_t)L_ * 2 * (4 * H2_) * H_);
  cvt(enc_Whh,  h_encWhh,  (size_t)L_ * 2 * (4 * H2_) * H2_);
  cvt(vdec_Wih, h_vdecWih, (size_t)L_ * (4 * H_) * H_);
  cvt(vdec_Whh, h_vdecWhh, (size_t)L_ * (4 * H_) * H_);
  cvt(dec_Wih0, h_decWih0, (size_t)(4 * H_) * (2 * H_));
  cvt(dec_Wih1, h_decWih1, (size_t)(4 * H_) * H_);
  cvt(dec_Whh,  h_decWhh,  (size_t)L_ * (4 * H_) * H_);
  cvt(vlinin,   h_vlinin,  (size_t)H_ * H_);
  cvt(vlinout,  h_vlinout, (size_t)H_ * 2 * H_);
  cvt(dvlinin,  h_dvlinin, (size_t)H_ * H_);
  cvt(linout,   h_linout,  (size_t)H_ * 2 * H_);
  cvt(gen_w,    h_gen,     (size_t)V_ * H_);
  cvt(vgen_w,   h_vgen,    (size_t)VV_ * H_);
  fillz(hzero, B_ * H_);

  // ---- bidirectional encoder ----
  gather(encemb, t_inp, xA, B_ * S_);
  float* xin = xA; float* xout = xB;
  for (int l = 0; l < L_; ++l) {
    for (int d = 0; d < 2; ++d) {
      const _Float16* Wih = h_encWih + (size_t)(l * 2 + d) * (4 * H2_) * H_;
      const _Float16* Whh = h_encWhh + (size_t)(l * 2 + d) * (4 * H2_) * H2_;
      const float* bg = enc_b + (size_t)(l * 2 + d) * (4 * H2_);
      // input-gate precompute for whole sequence: (B*S, 4*H2) = x @ Wih^T + b
      gemm(xin, H_, Wih, bg, pre, 4 * H2_, B_ * S_, 4 * H2_, H_, 0);
      fillz(cbuf, B_ * H2_);
      for (int st = 0; st < S_; ++st) {
        int t = d ? (S_ - 1 - st) : st;
        const float* hp; int lda;
        if (st == 0) { hp = hzero; lda = H_; }
        else { int tp = d ? (t + 1) : (t - 1); hp = xout + (size_t)tp * H_ + d * H2_; lda = S_ * H_; }
        gemm(hp, lda, Whh, nullptr, pre + (size_t)t * (4 * H2_), (size_t)S_ * (4 * H2_),
             B_, 4 * H2_, H2_, FLAG_ACC);
        cell(pre + (size_t)t * (4 * H2_), (size_t)S_ * (4 * H2_), cbuf,
             xout + (size_t)t * H_ + d * H2_, (size_t)S_ * H_, H2_);
      }
      int tf = d ? 0 : (S_ - 1);
      copy2d(xout + (size_t)tf * H_ + d * H2_, (size_t)S_ * H_,
             h0s + (size_t)l * B_ * H_ + d * H2_, H_, B_, H2_);
      copy2d(cbuf, H2_, c0s + (size_t)l * B_ * H_ + d * H2_, H_, B_, H2_);
    }
    float* tmp = xin; xin = xout; xout = tmp;
  }
  float* encp = xin;  // (B,S,H) encoder output

  // ---- verb decoder (runs twice from same (h0,c0)) ----
  auto run_vdec = [&](const float* embIn, float* outSeq) {
    const float* lin = embIn;
    for (int l = 0; l < L_; ++l) {
      float* lout = (l == 0) ? vyA : outSeq;
      const _Float16* Wih = h_vdecWih + (size_t)l * (4 * H_) * H_;
      const _Float16* Whh = h_vdecWhh + (size_t)l * (4 * H_) * H_;
      gemm(lin, H_, Wih, vdec_b + (size_t)l * (4 * H_), pre, 4 * H_, B_ * TV_, 4 * H_, H_, 0);
      copy2d(c0s + (size_t)l * B_ * H_, H_, cbuf, H_, B_, H_);
      for (int t = 0; t < TV_; ++t) {
        const float* hp = t ? (const float*)(lout + (size_t)(t - 1) * H_)
                            : (const float*)(h0s + (size_t)l * B_ * H_);
        int lda = t ? TV_ * H_ : H_;
        gemm(hp, lda, Whh, nullptr, pre + (size_t)t * (4 * H_), (size_t)TV_ * (4 * H_),
             B_, 4 * H_, H_, FLAG_ACC);
        cell(pre + (size_t)t * (4 * H_), (size_t)TV_ * (4 * H_), cbuf,
             lout + (size_t)t * H_, (size_t)TV_ * H_, H_);
      }
      lin = lout;
    }
  };

  // pass 1: teacher forced
  k_shift<<<(B_ * TV_ + 255) / 256, 256, 0, stream>>>(t_verbs, tokbuf, B_, TV_);
  gather(vemb, tokbuf, vembBuf, B_ * TV_);
  run_vdec(vembBuf, vyB);  // decout_v

  // verb attention + vgen
  gemm(vyB, H_, h_vlinin, nullptr, qbuf, H_, B_ * TV_, H_, H_, 0);
  { int wv = B_ * TV_ * S_;
    k_scores<<<(wv * 32 + 255) / 256, 256, 0, stream>>>(encp, qbuf, watt, B_, TV_, S_, H_); }
  k_softmax<<<(B_ * TV_ * 32 + 255) / 256, 256, 0, stream>>>(watt, B_ * TV_, S_);
  { int n = B_ * TV_ * H_;
    k_ctx<<<(n + 255) / 256, 256, 0, stream>>>(watt, encp, ccb, B_, TV_, S_, H_); }
  copy2d(ccb, H_, catb, 2 * H_, B_ * TV_, H_);
  copy2d(vyB, H_, catb + H_, 2 * H_, B_ * TV_, H_);
  gemm(catb, 2 * H_, h_vlinout, nullptr, opv, H_, B_ * TV_, H_, 2 * H_, FLAG_TANH);
  gemm(opv, H_, h_vgen, vgen_b, vout, VV_, B_ * TV_, VV_, H_, 0);
  k_argmax<<<(B_ * TV_ * 32 + 255) / 256, 256, 0, stream>>>(vout, B_ * TV_, VV_, vidxs);

  // pass 2: re-embed argmax verbs -> vencoding
  gather(vemb, vidxs, vembBuf, B_ * TV_);
  run_vdec(vembBuf, vyC);  // vencoding

  // ---- word decoder with input feeding ----
  k_shift<<<(B_ * TW_ + 255) / 256, 256, 0, stream>>>(t_out, tokbuf, B_, TW_);
  gather(decemb, tokbuf, demb, B_ * TW_);
  copy2d(h0s, H_, hw0, H_, B_, H_);
  copy2d(h0s + (size_t)B_ * H_, H_, hw1, H_, B_, H_);
  copy2d(c0s, H_, cw0, H_, B_, H_);
  copy2d(c0s + (size_t)B_ * H_, H_, cw1, H_, B_, H_);
  fillz(opb, B_ * H_);

  for (int t = 0; t < TW_; ++t) {
    // x = [emb_t | op]
    copy2d(demb + (size_t)t * H_, (size_t)TW_ * H_, xcat, 2 * H_, B_, H_);
    copy2d(opb, H_, xcat + H_, 2 * H_, B_, H_);
    // layer-0 cell
    gemm(xcat, 2 * H_, h_decWih0, dec_b, gbuf, 4 * H_, B_, 4 * H_, 2 * H_, 0);
    gemm(hw0, H_, h_decWhh, nullptr, gbuf, 4 * H_, B_, 4 * H_, H_, FLAG_ACC);
    { int n = B_ * H_;
      k_lstm_cell<<<(n + 255) / 256, 256, 0, stream>>>(gbuf, 4 * H_, cw0, cw0, hw0, H_, B_, H_); }
    // layer-1 cell
    gemm(hw0, H_, h_decWih1, dec_b + 4 * H_, gbuf, 4 * H_, B_, 4 * H_, H_, 0);
    gemm(hw1, H_, h_decWhh + (size_t)(4 * H_) * H_, nullptr, gbuf, 4 * H_, B_, 4 * H_, H_, FLAG_ACC);
    { int n = B_ * H_;
      k_lstm_cell<<<(n + 255) / 256, 256, 0, stream>>>(gbuf, 4 * H_, cw1, cw1, hw1, H_, B_, H_); }
    // verb attention over vencoding
    gemm(hw1, H_, h_dvlinin, nullptr, vq, H_, B_, H_, H_, 0);
    { int wv = B_ * TV_;
      k_scores<<<(wv * 32 + 255) / 256, 256, 0, stream>>>(vyC, vq, vw, B_, 1, TV_, H_); }
    k_softmax<<<(B_ * 32 + 255) / 256, 256, 0, stream>>>(vw, B_, TV_);
    { int n = B_ * H_;
      k_ctx<<<(n + 255) / 256, 256, 0, stream>>>(vw, vyC, vcc, B_, 1, TV_, H_); }
    // op = tanh([decout | vcc] @ linout^T)
    copy2d(hw1, H_, cat2, 2 * H_, B_, H_);
    copy2d(vcc, H_, cat2 + H_, 2 * H_, B_, H_);
    gemm(cat2, 2 * H_, h_linout, nullptr, opb, H_, B_, H_, 2 * H_, FLAG_TANH);
    // logits -> d_out[:, t, :]
    gemm(opb, H_, h_gen, gen_b, outp + (size_t)t * V_, (size_t)TW_ * V_, B_, V_, H_, 0);
  }
}
